// GRU_19086834664042
// MI455X (gfx1250) — matrix operand order repaired and checked by reading, not compiled, not run on hardware
//
#include <hip/hip_runtime.h>
#include <hip/hip_bf16.h>
#include <math.h>

// Problem dims
#define V_ 10000
#define E_ 256
#define H_ 512
#define L_ 2
#define S_ 256
#define B_ 64
#define NWG_REC 16   // persistent workgroups for the recurrence kernel (128 waves)

typedef _Float16 half_t;
typedef half_t v16h __attribute__((ext_vector_type(16)));
typedef half_t v8h  __attribute__((ext_vector_type(8)));
typedef float  v8f  __attribute__((ext_vector_type(8)));

// ---------------------------------------------------------------------------
// WMMA helpers: v_wmma_f32_16x16x32_f16, D = A(16x32) x B(32x16) + C
// ---------------------------------------------------------------------------
__device__ __forceinline__ v8f wmma32(v16h a, v16h b, v8f c) {
  return __builtin_amdgcn_wmma_f32_16x16x32_f16(
      /*neg_a=*/false, a, /*neg_b=*/false, b,
      /*c_mod=*/(short)0, c, /*reuse_a=*/false, /*reuse_b=*/false);
}

// A fragment (16x32 f16) from row-major matrix; p points at (row0, k0); ld in halves.
// ISA layout: lanes 0-15 row M=lane hold K 0-7 (v0-3) and 16-23 (v4-7);
//             lanes 16-31 row M=lane-16 hold K 8-15 and 24-31.
__device__ __forceinline__ v16h load_a(const half_t* __restrict__ p, int ld, int lane) {
  int r  = lane & 15;
  int ks = (lane >> 4) << 3;               // 0 or 8
  const half_t* q = p + (size_t)r * ld + ks;
  v8h lo = *(const v8h*)(q);               // K ks..ks+7
  v8h hi = *(const v8h*)(q + 16);          // K ks+16..ks+23
  return __builtin_shufflevector(lo, hi, 0,1,2,3,4,5,6,7,8,9,10,11,12,13,14,15);
}

// B fragment (32x16 f16) from row-major W^T [N][K]; p points at (n0 row, k0).
// ISA layout: lanes 0-15 col N=lane hold K 0-7 and 16-23; lanes 16-31 col N=lane-16 hold K 8-15 and 24-31.
__device__ __forceinline__ v16h load_b(const half_t* __restrict__ p, int ld, int lane) {
  int n  = lane & 15;
  int ks = (lane >> 4) << 3;               // 0 or 8
  const half_t* q = p + (size_t)n * ld + ks;
  v8h lo = *(const v8h*)(q);
  v8h hi = *(const v8h*)(q + 16);
  return __builtin_shufflevector(lo, hi, 0,1,2,3,4,5,6,7,8,9,10,11,12,13,14,15);
}

// A fragment for (r .* h): r is f32 [B][1024] (cols 0..511 = r), h is f16 [B][H]
__device__ __forceinline__ v16h load_rh(const float* __restrict__ rz,
                                        const half_t* __restrict__ h,
                                        int mbase, int k, int lane) {
  int row = mbase + (lane & 15);
  int ks  = k + ((lane >> 4) << 3);
  const float*  rp = rz + (size_t)row * 1024 + ks;
  const half_t* hp = h  + (size_t)row * H_   + ks;
  v16h o;
  #pragma unroll
  for (int i = 0; i < 8; ++i) o[i]     = (half_t)(rp[i]      * (float)hp[i]);
  #pragma unroll
  for (int i = 0; i < 8; ++i) o[8 + i] = (half_t)(rp[16 + i] * (float)hp[16 + i]);
  return o;
}

// ---------------------------------------------------------------------------
// Prep: f32 [K][N] -> f16 transposed [N][K]
// ---------------------------------------------------------------------------
__global__ void transpose_pack(const float* __restrict__ src, half_t* __restrict__ dst,
                               int K, int N) {
  long idx = (long)blockIdx.x * blockDim.x + threadIdx.x;
  if (idx >= (long)K * N) return;
  int  k = (int)(idx % K);
  long n = idx / K;
  dst[n * K + k] = (half_t)src[(long)k * N + n];
}

// Embedding gather -> f16 x0 [S][B][E]
__global__ void embed_kernel(const int* __restrict__ tok, const float* __restrict__ emb,
                             half_t* __restrict__ x0) {
  long idx = (long)blockIdx.x * blockDim.x + threadIdx.x;   // exact S*B*E
  int  e  = (int)(idx % E_);
  long tb = idx / E_;
  x0[idx] = (half_t)emb[(long)tok[tb] * E_ + e];
}

// Init: h slot0 <- f16(hidden), zero barrier counters
__global__ void init_kernel(const float* __restrict__ hidden,
                            half_t* __restrict__ h0s, half_t* __restrict__ h1s,
                            int* __restrict__ bar) {
  int idx = blockIdx.x * blockDim.x + threadIdx.x;
  if (idx < B_ * H_) {
    h0s[idx] = (half_t)hidden[idx];
    h1s[idx] = (half_t)hidden[B_ * H_ + idx];
  }
  if (blockIdx.x == 0 && threadIdx.x < 8) bar[threadIdx.x] = 0;
}

// ---------------------------------------------------------------------------
// X-part GEMM, register-blocked 2x4 (pressure-safe):
//   G[M][1536] = A[M][kx]f16 @ Wt(gate)[512][ldw]^T (K in [0,kx)) + bias
// M = S*B = 16384 -> 512 M-groups of 32 rows; N: 96 tiles -> 24 groups of 4
// (gate boundary at tile 32/64 is 4-aligned -> a group never straddles gates)
// waves = 512*24 = 12288 -> 1536 blocks x 8 waves
// ---------------------------------------------------------------------------
__global__ void __launch_bounds__(256, 1) xgemm_kernel(
    const half_t* __restrict__ A, int lda, int kx,
    const half_t* __restrict__ Wt, int ldw,
    const float* __restrict__ b0, const float* __restrict__ b1,
    const float* __restrict__ b2, float* __restrict__ G) {
  int lane = threadIdx.x & 31;
  int wave = (int)(((long)blockIdx.x * blockDim.x + threadIdx.x) >> 5);
  int mg = wave / 24, ng = wave % 24;
  int ntile0 = ng * 4;
  int gate = ntile0 >> 5, ncol0 = (ntile0 & 31) << 4;
  const half_t* arow0 = A  + (size_t)(mg * 32) * lda;
  const half_t* brow0 = Wt + (size_t)(gate * 512 + ncol0) * ldw;
  v8f acc[2][4] = {};
  for (int k = 0; k < kx; k += 32) {
    v16h a0 = load_a(arow0 + k, lda, lane);
    v16h a1 = load_a(arow0 + (size_t)16 * lda + k, lda, lane);
    #pragma unroll
    for (int j = 0; j < 4; ++j) {
      v16h b = load_b(brow0 + (size_t)(j * 16) * ldw + k, ldw, lane);
      acc[0][j] = wmma32(a0, b, acc[0][j]);
      acc[1][j] = wmma32(a1, b, acc[1][j]);
    }
  }
  const float* bias = (gate == 0) ? b0 : ((gate == 1) ? b1 : b2);
  int hi = lane >> 4, nn = lane & 15;
  #pragma unroll
  for (int j = 0; j < 4; ++j) {
    float bv = bias[ncol0 + j * 16 + nn];
    #pragma unroll
    for (int i = 0; i < 2; ++i) {
      float* gout = G + (size_t)(mg * 32 + i * 16) * 1536 + gate * 512 + ncol0 + j * 16 + nn;
      #pragma unroll
      for (int r = 0; r < 8; ++r) gout[(size_t)(r + 8 * hi) * 1536] = acc[i][j][r] + bv;
    }
  }
}

// ---------------------------------------------------------------------------
// Global generation barrier over NWG_REC workgroups
// ---------------------------------------------------------------------------
__device__ __forceinline__ void grid_barrier(int* bar) {
  __syncthreads();
  if (threadIdx.x == 0) {
    __threadfence();
    int gen  = __hip_atomic_load(bar + 1, __ATOMIC_RELAXED, __HIP_MEMORY_SCOPE_AGENT);
    int prev = __hip_atomic_fetch_add(bar + 0, 1, __ATOMIC_ACQ_REL, __HIP_MEMORY_SCOPE_AGENT);
    if (prev == NWG_REC - 1) {
      __hip_atomic_store(bar + 0, 0, __ATOMIC_RELAXED, __HIP_MEMORY_SCOPE_AGENT);
      __hip_atomic_fetch_add(bar + 1, 1, __ATOMIC_RELEASE, __HIP_MEMORY_SCOPE_AGENT);
    } else {
      while (__hip_atomic_load(bar + 1, __ATOMIC_ACQUIRE, __HIP_MEMORY_SCOPE_AGENT) == gen)
        __builtin_amdgcn_s_sleep(2);
    }
  }
  __syncthreads();
}

// ---------------------------------------------------------------------------
// Persistent GRU recurrence for one layer.
//   phase1: rz = sigmoid(h_{t-1} @ [Wr_h|Wz_h] + Gx_rz)
//           each wave: one (mi, ncol) with BOTH r and z gates sharing the A frag
//   phase2: c = tanh((r.*h) @ Wh_h + Gx_h); h_t = (1-z)h + z c   (128 tiles)
// Wt = [3][512][ldw] transposed f16 combined weights; h-part K offset = inSz.
// ---------------------------------------------------------------------------
__global__ void __launch_bounds__(256) gru_rec_kernel(
    const half_t* __restrict__ Wt, int ldw, int inSz,
    const float* __restrict__ G, half_t* __restrict__ hseq,
    float* __restrict__ rz, float* __restrict__ hfinal, int* bar) {
  int lane = threadIdx.x & 31;
  int gw   = (blockIdx.x * blockDim.x + threadIdx.x) >> 5;  // 0..127
  int hi = lane >> 4, nn = lane & 15;
  int mi = gw >> 5, ncol = (gw & 31) << 4;                  // shared tile coords

  for (int t = 0; t < S_; ++t) {
    const half_t* hprev = hseq + (size_t)t * B_ * H_;
    const float*  Gt    = G    + (size_t)t * B_ * 1536;

    // phase 1: r and z gate tiles for (mi, ncol); A fragment loaded once
    {
      const half_t* arow  = hprev + (size_t)(mi * 16) * H_;
      const half_t* browR = Wt + (size_t)(ncol) * ldw + inSz;
      const half_t* browZ = Wt + (size_t)(512 + ncol) * ldw + inSz;
      v8f accR = {}, accZ = {};
      for (int k = 0; k < H_; k += 32) {
        v16h a = load_a(arow + k, H_, lane);
        accR = wmma32(a, load_b(browR + k, ldw, lane), accR);
        accZ = wmma32(a, load_b(browZ + k, ldw, lane), accZ);
      }
      #pragma unroll
      for (int i = 0; i < 8; ++i) {
        int b = mi * 16 + i + 8 * hi;
        int n = ncol + nn;
        float vr = accR[i] + Gt[(size_t)b * 1536 + n];
        float vz = accZ[i] + Gt[(size_t)b * 1536 + 512 + n];
        rz[(size_t)b * 1024 + n]       = 1.0f / (1.0f + expf(-vr));
        rz[(size_t)b * 1024 + 512 + n] = 1.0f / (1.0f + expf(-vz));
      }
    }
    grid_barrier(bar);

    // phase 2: candidate + state update (one tile per wave)
    {
      const half_t* brow = Wt + (size_t)(2 * 512 + ncol) * ldw + inSz;
      v8f acc = {};
      for (int k = 0; k < H_; k += 32)
        acc = wmma32(load_rh(rz, hprev, mi * 16, k, lane),
                     load_b(brow + k, ldw, lane), acc);
      half_t* hnext = hseq + (size_t)(t + 1) * B_ * H_;
      #pragma unroll
      for (int i = 0; i < 8; ++i) {
        int b = mi * 16 + i + 8 * hi;
        int n = ncol + nn;
        float c  = tanhf(acc[i] + Gt[(size_t)b * 1536 + 1024 + n]);
        float z  = rz[(size_t)b * 1024 + 512 + n];
        float hp = (float)hprev[(size_t)b * H_ + n];
        float hn = (1.0f - z) * hp + z * c;
        hnext[(size_t)b * H_ + n] = (half_t)hn;
        if (t == S_ - 1) hfinal[(size_t)b * H_ + n] = hn;
      }
    }
    grid_barrier(bar);
  }
}

// ---------------------------------------------------------------------------
// Logits GEMM, register-blocked 2x5 (pressure-safe; 625 N-tiles = 125 x 5):
//   out[16384][10000] = h1[16384][512] @ WoutT[10000][512]^T + bout
// M: 1024 tiles -> 512 groups of 2; N: 625 tiles -> 125 groups of 5
// waves = 512*125 = 64000 -> 8000 blocks x 8 waves
// ---------------------------------------------------------------------------
__global__ void __launch_bounds__(256, 1) logits_kernel(
    const half_t* __restrict__ A, const half_t* __restrict__ WtO,
    const float* __restrict__ bout, float* __restrict__ out) {
  int  lane = threadIdx.x & 31;
  long wave = ((long)blockIdx.x * blockDim.x + threadIdx.x) >> 5;
  const int NG = 125;
  int mg = (int)(wave / NG), ng = (int)(wave % NG);
  const half_t* arow0 = A   + (size_t)(mg * 32) * H_;
  const half_t* brow0 = WtO + (size_t)(ng * 80) * H_;
  v8f acc[2][5] = {};
  for (int k = 0; k < H_; k += 32) {
    v16h a0 = load_a(arow0 + k, H_, lane);
    v16h a1 = load_a(arow0 + (size_t)16 * H_ + k, H_, lane);
    #pragma unroll
    for (int j = 0; j < 5; ++j) {
      v16h b = load_b(brow0 + (size_t)(j * 16) * H_ + k, H_, lane);
      acc[0][j] = wmma32(a0, b, acc[0][j]);
      acc[1][j] = wmma32(a1, b, acc[1][j]);
    }
  }
  int hi = lane >> 4, nn = lane & 15;
  #pragma unroll
  for (int j = 0; j < 5; ++j) {
    float bv = bout[ng * 80 + j * 16 + nn];
    #pragma unroll
    for (int i = 0; i < 2; ++i) {
      float* orow = out + (size_t)(mg * 32 + i * 16) * V_ + ng * 80 + j * 16 + nn;
      #pragma unroll
      for (int r = 0; r < 8; ++r) orow[(size_t)(r + 8 * hi) * V_] = acc[i][j][r] + bv;
    }
  }
}

// ---------------------------------------------------------------------------
extern "C" void kernel_launch(void* const* d_in, const int* in_sizes, int n_in,
                              void* d_out, int out_size, void* d_ws, size_t ws_size,
                              hipStream_t stream) {
  const int*   tok    = (const int*)d_in[0];
  const float* hidden = (const float*)d_in[1];
  const float* emb    = (const float*)d_in[2];
  const float* Wr0 = (const float*)d_in[3];  const float* br0 = (const float*)d_in[4];
  const float* Wz0 = (const float*)d_in[5];  const float* bz0 = (const float*)d_in[6];
  const float* Wh0 = (const float*)d_in[7];  const float* bh0 = (const float*)d_in[8];
  const float* Wr1 = (const float*)d_in[9];  const float* br1 = (const float*)d_in[10];
  const float* Wz1 = (const float*)d_in[11]; const float* bz1 = (const float*)d_in[12];
  const float* Wh1 = (const float*)d_in[13]; const float* bh1 = (const float*)d_in[14];
  const float* Wout = (const float*)d_in[15]; const float* bout = (const float*)d_in[16];
  float* out = (float*)d_out;

  // workspace carve-up (256B aligned)
  char* w = (char*)d_ws;
  auto carve = [&](size_t bytes) { char* p = w; w += (bytes + 255) & ~(size_t)255; return p; };
  half_t* Wt0 = (half_t*)carve((size_t)3 * 512 * 768 * 2);
  half_t* Wt1 = (half_t*)carve((size_t)3 * 512 * 1024 * 2);
  half_t* WtO = (half_t*)carve((size_t)V_ * H_ * 2);
  half_t* x0  = (half_t*)carve((size_t)S_ * B_ * E_ * 2);
  half_t* h0  = (half_t*)carve((size_t)(S_ + 1) * B_ * H_ * 2);
  half_t* h1  = (half_t*)carve((size_t)(S_ + 1) * B_ * H_ * 2);
  float*  G   = (float*)carve((size_t)S_ * B_ * 1536 * 4);   // reused by both layers
  float*  rz  = (float*)carve((size_t)B_ * 1024 * 4);
  int*    bar = (int*)carve(64);

  // 1) weight prep: f32 [K][N] -> f16 [N][K]
  auto tp = [&](const float* src, half_t* dst, int K, int N) {
    long tot = (long)K * N;
    transpose_pack<<<(int)((tot + 255) / 256), 256, 0, stream>>>(src, dst, K, N);
  };
  tp(Wr0, Wt0,                 768, 512);
  tp(Wz0, Wt0 + 512 * 768,     768, 512);
  tp(Wh0, Wt0 + 2 * 512 * 768, 768, 512);
  tp(Wr1, Wt1,                  1024, 512);
  tp(Wz1, Wt1 + 512 * 1024,     1024, 512);
  tp(Wh1, Wt1 + 2 * 512 * 1024, 1024, 512);
  tp(Wout, WtO, 512, V_);

  init_kernel<<<(B_ * H_ + 255) / 256, 256, 0, stream>>>(hidden, h0, h1, bar);
  embed_kernel<<<S_ * B_ * E_ / 256, 256, 0, stream>>>(tok, emb, x0);

  float* hfinal = out + (size_t)S_ * B_ * V_;

  // layer 0: x-part GEMM (K=E), then persistent recurrence
  xgemm_kernel<<<1536, 256, 0, stream>>>(x0, E_, E_, Wt0, 768, br0, bz0, bh0, G);
  gru_rec_kernel<<<NWG_REC, 256, 0, stream>>>(Wt0, 768, E_, G, h0, rz, hfinal, bar);

  // layer 1: x = h0[t] (K=H), then recurrence
  xgemm_kernel<<<1536, 256, 0, stream>>>(h0 + B_ * H_, H_, H_, Wt1, 1024, br1, bz1, bh1, G);
  gru_rec_kernel<<<NWG_REC, 256, 0, stream>>>(Wt1, 1024, H_, G, h1, rz, hfinal + B_ * H_, bar);

  // output projection for all timesteps
  logits_kernel<<<8000, 256, 0, stream>>>(h1 + B_ * H_, WtO, bout, out);
}
